// WL2Layer_34651796144208
// MI455X (gfx1250) — matrix operand order repaired and checked by reading, not compiled, not run on hardware
//
#include <hip/hip_runtime.h>

typedef _Float16 v16h __attribute__((ext_vector_type(16)));
typedef _Float16 v8h  __attribute__((ext_vector_type(8)));
typedef float    v8f  __attribute__((ext_vector_type(8)));

#define N_ROWS   200000   // multiple of 16: each 16-row wave tile is all-valid or all-invalid
#define XDIM     128
#define UNITS    128
#define N_EDGES  2000000
#define BT_PITCH 136      // halfs per transposed-W row in LDS (8-half aligned, de-conflicted)

// ---------------------------------------------------------------------------
// Phase 1: fused triple GEMM  C_w = X @ W_w   (f16 WMMA, f32 accumulate)
// ---------------------------------------------------------------------------
__device__ __forceinline__ void cvt4(v16h& a, int base, float4 f) {
  a[base + 0] = (_Float16)f.x; a[base + 1] = (_Float16)f.y;
  a[base + 2] = (_Float16)f.z; a[base + 3] = (_Float16)f.w;
}

__device__ __forceinline__ void gemm_one(const float* __restrict__ W,
                                         float* __restrict__ C,
                                         const v16h (&a)[4],
                                         _Float16* BT,
                                         int tid, int m, int half, int rowBase)
{
  __syncthreads();
  // Stage W transposed into LDS as f16: BT[n][k]
  for (int i = tid; i < XDIM * UNITS; i += 256) {
    int k = i >> 7;          // W is [K=128][N=128] row-major
    int n = i & 127;
    BT[n * BT_PITCH + k] = (_Float16)W[i];
  }
  __syncthreads();

  // Wave-uniform validity: N_ROWS % 16 == 0, so a 16-row tile starting below
  // N_ROWS is fully in range.
  const bool storeOk = (rowBase < N_ROWS);

  #pragma unroll
  for (int nt = 0; nt < 8; ++nt) {           // 8 column tiles of 16
    v8f acc = {};
    #pragma unroll
    for (int s = 0; s < 4; ++s) {            // K = 128 in 4 steps of 32
      // B layout: lane L -> N = L%16, halfs j=0..7 -> K = s*32 + (L/16)*8 + j, halfs 8..15 -> K = s*32 + 16 + (L/16)*8 + j
      int off = (nt * 16 + m) * BT_PITCH + s * 32 + half * 8;
      v8h blo = *(const v8h*)(&BT[off]);
      v8h bhi = *(const v8h*)(&BT[off + 16]);
      v16h b;
      #pragma unroll
      for (int j = 0; j < 8; ++j) { b[j] = blo[j]; b[8 + j] = bhi[j]; }
      acc = __builtin_amdgcn_wmma_f32_16x16x32_f16(
          /*neg_a=*/false, a[s], /*neg_b=*/false, b,
          /*c_mod=*/(short)0, acc, /*reuse_a=*/false, /*reuse_b=*/false);
    }
    // C layout: VGPR r -> lanes 0-15: M=r, lanes 16-31: M=r+8; N = lane%16
    if (storeOk) {
      float* cp = C + (long)(rowBase + half * 8) * UNITS + nt * 16 + m;
      #pragma unroll
      for (int r = 0; r < 8; ++r) cp[r * UNITS] = acc[r];   // imm offset r*512B
    }
  }
}

__global__ __launch_bounds__(256) void gemm3_kernel(
    const float* __restrict__ X,
    const float* __restrict__ W0, const float* __restrict__ W1,
    const float* __restrict__ W2,
    float* __restrict__ C0, float* __restrict__ C1, float* __restrict__ C2)
{
  __shared__ _Float16 BT[UNITS * BT_PITCH];   // 34.0 KB
  const int tid     = threadIdx.x;
  const int lane    = tid & 31;
  const int wave    = tid >> 5;               // 8 waves/block (wave32)
  const int m       = lane & 15;
  const int half    = lane >> 4;
  const int rowBase = blockIdx.x * 128 + wave * 16;

  // A fragments: 16 rows x K=128 per wave, converted f32->f16 once.
  // A layout: lane L -> M = L%16; halfs 0..7  -> K = kb + (L/16)*8 + j
  //                               halfs 8..15 -> K = kb + 16 + (L/16)*8 + j
  const int  rowA = rowBase + m;
  const long rCl  = (rowA < N_ROWS) ? rowA : (N_ROWS - 1);   // clamp (stores guarded)
  const float* xr = X + rCl * XDIM;
  v16h a[4];
  #pragma unroll
  for (int s = 0; s < 4; ++s) {
    const float4* p0 = (const float4*)(xr + 32 * s + half * 8);
    const float4* p1 = (const float4*)(xr + 32 * s + 16 + half * 8);
    cvt4(a[s], 0,  p0[0]); cvt4(a[s], 4,  p0[1]);
    cvt4(a[s], 8,  p1[0]); cvt4(a[s], 12, p1[1]);
  }

  gemm_one(W0, C0, a, BT, tid, m, half, rowBase);   // XW_local  -> d_out
  gemm_one(W1, C1, a, BT, tid, m, half, rowBase);   // XW_filter -> ws
  gemm_one(W2, C2, a, BT, tid, m, half, rowBase);   // XW_neighbor -> ws
}

// ---------------------------------------------------------------------------
// Zero the X_conv accumulator (must not rely on prior buffer state)
// ---------------------------------------------------------------------------
__global__ __launch_bounds__(256) void zero_kernel(float4* __restrict__ p, long n4)
{
  long i = (long)blockIdx.x * blockDim.x + threadIdx.x;
  if (i < n4) p[i] = make_float4(0.f, 0.f, 0.f, 0.f);
}

// ---------------------------------------------------------------------------
// Phase 2: msg = relu(XWn[a] + XWn[b] + b_n); X_conv[backref] += msg
// One wave per edge, 4 units per lane.
// ---------------------------------------------------------------------------
__global__ __launch_bounds__(256) void edge_kernel(
    const float* __restrict__ XWn,
    const int* __restrict__ ra, const int* __restrict__ rb,
    const int* __restrict__ br,
    const float* __restrict__ bn,
    float* __restrict__ conv)
{
  long t = (long)blockIdx.x * blockDim.x + threadIdx.x;
  long e = t >> 5;
  if (e >= N_EDGES) return;
  int c = (int)(t & 31) * 4;

  const float4 va = *(const float4*)(XWn + (long)ra[e] * UNITS + c);
  const float4 vb = *(const float4*)(XWn + (long)rb[e] * UNITS + c);
  const float4 bb = *(const float4*)(bn + c);

  float4 msg;
  msg.x = fmaxf(va.x + vb.x + bb.x, 0.f);
  msg.y = fmaxf(va.y + vb.y + bb.y, 0.f);
  msg.z = fmaxf(va.z + vb.z + bb.z, 0.f);
  msg.w = fmaxf(va.w + vb.w + bb.w, 0.f);

  float* d = conv + (long)br[e] * UNITS + c;
  atomicAdd(d + 0, msg.x);
  atomicAdd(d + 1, msg.y);
  atomicAdd(d + 2, msg.z);
  atomicAdd(d + 3, msg.w);
}

// ---------------------------------------------------------------------------
// Phase 3: out = relu(XW_local + XW_filter * X_conv + b)   (in place on d_out)
// ---------------------------------------------------------------------------
__global__ __launch_bounds__(256) void final_kernel(
    float* __restrict__ out,
    const float* __restrict__ XWf,
    const float* __restrict__ conv,
    const float* __restrict__ b)
{
  long i = (long)blockIdx.x * blockDim.x + threadIdx.x;   // float4 index
  if (i >= (long)N_ROWS * UNITS / 4) return;
  int c = (int)((i * 4) & (UNITS - 1));

  float4 l  = ((const float4*)out)[i];
  float4 f  = ((const float4*)XWf)[i];
  float4 v  = ((const float4*)conv)[i];
  float4 bb = *(const float4*)(b + c);

  float4 r;
  r.x = fmaxf(l.x + f.x * v.x + bb.x, 0.f);
  r.y = fmaxf(l.y + f.y * v.y + bb.y, 0.f);
  r.z = fmaxf(l.z + f.z * v.z + bb.z, 0.f);
  r.w = fmaxf(l.w + f.w * v.w + bb.w, 0.f);
  ((float4*)out)[i] = r;
}

// ---------------------------------------------------------------------------
extern "C" void kernel_launch(void* const* d_in, const int* in_sizes, int n_in,
                              void* d_out, int out_size, void* d_ws, size_t ws_size,
                              hipStream_t stream)
{
  const float* X  = (const float*)d_in[0];
  const int*   ra = (const int*)d_in[1];
  const int*   rb = (const int*)d_in[2];
  const int*   br = (const int*)d_in[3];
  const float* Wl = (const float*)d_in[4];
  const float* Wf = (const float*)d_in[5];
  const float* Wn = (const float*)d_in[6];
  const float* b  = (const float*)d_in[7];
  const float* bn = (const float*)d_in[8];

  float* out = (float*)d_out;

  const size_t mat = (size_t)N_ROWS * UNITS;   // 25.6M floats
  float* wsFilter   = (float*)d_ws;            // XW_filter
  float* wsNeighbor = wsFilter + mat;          // XW_neighbor
  float* wsConv     = wsNeighbor + mat;        // X_conv accumulator

  // Phase 1: 128 rows per block
  int gemmBlocks = (N_ROWS + 127) / 128;       // 1563
  gemm3_kernel<<<gemmBlocks, 256, 0, stream>>>(X, Wl, Wf, Wn,
                                               out, wsFilter, wsNeighbor);

  // Zero X_conv (6.4M float4)
  long n4 = (long)mat / 4;
  zero_kernel<<<(int)((n4 + 255) / 256), 256, 0, stream>>>((float4*)wsConv, n4);

  // Phase 2: one wave per edge
  long edgeThreads = (long)N_EDGES * 32;
  edge_kernel<<<(int)((edgeThreads + 255) / 256), 256, 0, stream>>>(
      wsNeighbor, ra, rb, br, bn, wsConv);

  // Phase 3: elementwise combine
  final_kernel<<<(int)((n4 + 255) / 256), 256, 0, stream>>>(out, wsFilter, wsConv, b);
}
